// BasicBlock_15461882266334
// MI455X (gfx1250) — matrix operand order repaired and checked by reading, not compiled, not run on hardware
//
#include <hip/hip_runtime.h>

typedef __attribute__((ext_vector_type(16))) _Float16 v16h;
typedef __attribute__((ext_vector_type(8)))  _Float16 v8h;
typedef __attribute__((ext_vector_type(8)))  float    v8f;

#define BATCH   64
#define CCH     256
#define HGT     28
#define WID     28
#define HW      (HGT*WID)            // 784
#define NPOS    (BATCH*HW)           // 50176
#define KDIM    (CCH*9)              // 2304
#define WELEMS  (CCH*KDIM)           // 589824

#define TILE_M  128
#define TILE_N  128
#define TILE_K  32
#define KTILES  (KDIM/TILE_K)        // 72
#define LDS_PITCH 40                 // halves: 80B pitch = 64B data + 16B pad

#if defined(__HIP_DEVICE_COMPILE__) && __has_builtin(__builtin_amdgcn_tensor_load_to_lds)
#define USE_TDM 1
typedef unsigned int v4u  __attribute__((ext_vector_type(4)));
typedef int          v8ii __attribute__((ext_vector_type(8)));
typedef int          v4ii __attribute__((ext_vector_type(4)));
#else
#define USE_TDM 0
#endif

// ---------------------------------------------------------------------------
// Step 1: abs-max of each weight tensor (step = max|w| / 127)
// ---------------------------------------------------------------------------
__global__ void init_absmax(unsigned* p) {
  if (threadIdx.x < 2) p[threadIdx.x] = 0u;
}

__global__ void absmax_kernel(const float* __restrict__ w, unsigned* out, int n) {
  unsigned m = 0u;
  for (int i = blockIdx.x * blockDim.x + threadIdx.x; i < n;
       i += gridDim.x * blockDim.x) {
    unsigned b = __float_as_uint(fabsf(w[i]));   // monotonic for |w| >= 0
    m = m > b ? m : b;
  }
  atomicMax(out, m);
}

// ---------------------------------------------------------------------------
// Step 2: quantize weights to "levels" (dev + cent), stored f16 (exact:
// levels lie on a 0.125 grid with |v| <= 127, inside fp16's integer grid).
// Grain = 1 x 9 over [O, I*9]  ==> one (o,i) 3x3 filter per grain.
// ---------------------------------------------------------------------------
__global__ void quant_kernel(const float* __restrict__ w,
                             const unsigned* __restrict__ absbits,
                             _Float16* __restrict__ wq) {
  int g = blockIdx.x * blockDim.x + threadIdx.x;   // grain id, 0..65535
  if (g >= CCH * CCH) return;
  const float step     = __uint_as_float(*absbits) * (1.0f / 127.0f);
  const float inv_step = 1.0f / step;
  const float* src = w + (size_t)g * 9;
  float wl[9];
  float s = 0.0f;
#pragma unroll
  for (int j = 0; j < 9; ++j) {
    float t = rintf(src[j] * inv_step);
    t = fminf(fmaxf(t, -127.0f), 127.0f);
    wl[j] = t;
    s += t;
  }
  const float gm   = s * (1.0f / 9.0f);
  const float cstp = 127.0f / 8.0f;                 // 15.875
  const float cent = fminf(fmaxf(rintf(gm / cstp), -8.0f), 8.0f) * cstp;
  _Float16* dst = wq + (size_t)g * 9;
#pragma unroll
  for (int j = 0; j < 9; ++j) {
    float dev = rintf(fminf(fmaxf(wl[j] - cent, -63.5f), 63.5f));
    dst[j] = (_Float16)(dev + cent);
  }
}

// ---------------------------------------------------------------------------
// Step 3: x (f32 NCHW) -> f16 channel-major [C][B*H*W]
// ---------------------------------------------------------------------------
__global__ void nchw_to_cn_f16(const float* __restrict__ in,
                               _Float16* __restrict__ out) {
  size_t idx = (size_t)blockIdx.x * blockDim.x + threadIdx.x;
  if (idx >= (size_t)BATCH * CCH * HW) return;
  int yx = (int)(idx % HW);
  int bc = (int)(idx / HW);
  int c  = bc % CCH;
  int b  = bc / CCH;
  out[(size_t)c * NPOS + (size_t)b * HW + yx] = (_Float16)in[idx];
}

// ---------------------------------------------------------------------------
// Step 4: implicit-GEMM conv3x3 + BN (+residual), v_wmma_f32_16x16x32_f16.
//   M=256 (out ch), N=50176 (B*H*W), K=2304 (C*9).
//   256 threads = 8 waves (2x4); block tile 128x128; wave tile 64x32.
//   Double-buffered LDS; A (weight) tile via Tensor Data Mover with
//   pad_interval/pad_amount producing the 80B LDS pitch; B tile gathered
//   with hoisted n->(b,y,x) decomposition.
// ---------------------------------------------------------------------------
template <int STAGE>
__global__ __launch_bounds__(256)
void conv_bn_wmma(const _Float16* __restrict__ wq,     // [256][2304] f16 levels
                  const _Float16* __restrict__ act,    // [256][50176] f16
                  const unsigned* __restrict__ absbits,
                  const float* __restrict__ gamma, const float* __restrict__ beta,
                  const float* __restrict__ mean,  const float* __restrict__ var,
                  const float* __restrict__ resid,     // stage2: x, f32 NCHW
                  _Float16* __restrict__ out_h,        // stage1 destination
                  float* __restrict__ out_f) {         // stage2 destination
  __shared__ __align__(128) _Float16 As[2][TILE_M * LDS_PITCH];
  __shared__ __align__(128) _Float16 Bs[2][TILE_N * LDS_PITCH];

  const int t    = threadIdx.x;
  const int lane = t & 31;
  const int wv   = t >> 5;       // wave 0..7
  const int wm   = wv >> 2;      // 0..1  (M direction)
  const int wn   = wv & 3;       // 0..3  (N direction)
  const int half = lane >> 4;    // lane group 0/1 (ISA 16-bit A/B layout)
  const int lm   = lane & 15;

  const int n0 = blockIdx.x * TILE_N;
  const int m0 = blockIdx.y * TILE_M;

  v8f acc[4][2];
#pragma unroll
  for (int i = 0; i < 4; ++i)
#pragma unroll
    for (int j = 0; j < 2; ++j)
#pragma unroll
      for (int e = 0; e < 8; ++e) acc[i][j][e] = 0.0f;

#if !USE_TDM
  const int arow = t >> 1;           // A fallback loader mapping
  const int akh  = (t & 1) * 16;
#endif
  const int krow = t >> 3;           // B gather: k-row 0..31
  const int nset = t & 7;            // B gather: 16 n's per thread

  // ---- hoist loop-invariant n -> (b,y,x) decomposition ----
  int pos[16];   // b*HW + y*WID + x
  int pyx[16];   // (y<<8)|x
#pragma unroll
  for (int j = 0; j < 16; ++j) {
    const int n  = n0 + nset * 16 + j;
    const int b  = n / HW;
    const int yx = n - b * HW;
    const int y  = yx / WID;
    const int x  = yx - y * WID;
    pos[j] = b * HW + yx;
    pyx[j] = (y << 8) | x;
  }

  // ---- A tile loader: Tensor Data Mover (wave 0 issues) or b128 fallback --
  auto loadA = [&](int kt, int buf) {
#if USE_TDM
    if (wv == 0) {
      const unsigned lds_off =
          (unsigned)(uintptr_t)(&As[buf][0]);            // LDS byte offset
      const unsigned long long ga =
          (unsigned long long)(uintptr_t)(wq + (size_t)m0 * KDIM + kt * TILE_K);
      v4u g0;
      g0[0] = 1u;                                        // count = 1 (valid)
      g0[1] = lds_off;                                   // lds_addr
      g0[2] = (unsigned)ga;                              // global_addr[31:0]
      g0[3] = (unsigned)((ga >> 32) & 0x1ffffffu) | (2u << 30); // addr hi|type=2
      v8ii g1;
      g1[0] = (1 << 16)      // data_size = 2 bytes
            | (1 << 20)      // pad_enable
            | (3 << 22)      // pad_interval: 16 DWORDs (64B) of data
            | (3 << 25);     // pad_amount:   4 DWORDs (16B) of pad -> 80B pitch
      g1[1] = (int)(KDIM << 16);      // tensor_dim0[15:0]   (2304)
      g1[2] = (int)(CCH << 16);       // tensor_dim1[15:0]   (256 rows)
      g1[3] = (int)(TILE_K << 16);    // tile_dim0 = 32 elements (64B rows)
      g1[4] = TILE_M;                 // tile_dim1 = 128 rows
      g1[5] = KDIM;                   // tensor_dim0_stride[31:0] = 2304
      g1[6] = 0;
      g1[7] = 0;
      v4ii z4 = {0, 0, 0, 0};
      v8ii z8 = {0, 0, 0, 0, 0, 0, 0, 0};
      __builtin_amdgcn_tensor_load_to_lds(g0, g1, z4, z4, z8, 0);
    }
#else
    const uint4* g = (const uint4*)(wq + (size_t)(m0 + arow) * KDIM +
                                    kt * TILE_K + akh);
    uint4* s = (uint4*)(&As[buf][arow * LDS_PITCH + akh]);
    s[0] = g[0];
    s[1] = g[1];
#endif
  };

  // ---- B tile gather: per-iteration only k -> (c,ky,kx) decode ----
  auto gatherB = [&](int kt, int buf) {
    const int k   = kt * TILE_K + krow;
    const int c   = k / 9;
    const int r9  = k - c * 9;
    const int ky  = r9 / 3 - 1;
    const int kx  = r9 - (r9 / 3) * 3 - 1;
    const _Float16* base = act + (size_t)c * NPOS + (ky * WID + kx);
    _Float16* dst = &Bs[buf][(nset * 16) * LDS_PITCH + krow];
#pragma unroll
    for (int j = 0; j < 16; ++j) {
      const int y = pyx[j] >> 8;
      const int x = pyx[j] & 255;
      _Float16 v = (_Float16)0.0f;
      if ((unsigned)(y + ky) < (unsigned)HGT &&
          (unsigned)(x + kx) < (unsigned)WID)
        v = base[pos[j]];
      dst[j * LDS_PITCH] = v;
    }
  };

  // ---- software-pipelined main loop (ping-pong LDS) ----
  loadA(0, 0);
  gatherB(0, 0);
  int p = 0;
  for (int kt = 0; kt < KTILES; ++kt) {
#if USE_TDM
    if (wv == 0) __builtin_amdgcn_s_wait_tensorcnt(0);   // A[p] landed in LDS
#endif
    __syncthreads();                                     // B[p] visible too
    if (kt + 1 < KTILES) {                               // prefetch next tile
      loadA(kt + 1, p ^ 1);
      gatherB(kt + 1, p ^ 1);
    }

    // ---- fragments per ISA 7.12.2 lane layouts ----
    v16h afrag[4], bfrag[2];
#pragma unroll
    for (int mi = 0; mi < 4; ++mi) {
      // A 16x32: lane m holds K[half*8 .. +7] and K[16+half*8 .. +7]
      const _Float16* ap =
          &As[p][(wm * 64 + mi * 16 + lm) * LDS_PITCH + half * 8];
      v8h lo = *(const v8h*)ap;
      v8h hi = *(const v8h*)(ap + 16);
#pragma unroll
      for (int e = 0; e < 8; ++e) { afrag[mi][e] = lo[e]; afrag[mi][e + 8] = hi[e]; }
    }
#pragma unroll
    for (int ni = 0; ni < 2; ++ni) {
      // B 32x16: lane n holds K[half*8 .. +7] and K[16+half*8 .. +7] of col n
      const _Float16* bp =
          &Bs[p][(wn * 32 + ni * 16 + lm) * LDS_PITCH + half * 8];
      v8h lo = *(const v8h*)bp;
      v8h hi = *(const v8h*)(bp + 16);
#pragma unroll
      for (int e = 0; e < 8; ++e) { bfrag[ni][e] = lo[e]; bfrag[ni][e + 8] = hi[e]; }
    }

#pragma unroll
    for (int mi = 0; mi < 4; ++mi)
#pragma unroll
      for (int ni = 0; ni < 2; ++ni)
        acc[mi][ni] = __builtin_amdgcn_wmma_f32_16x16x32_f16(
            false, afrag[mi], false, bfrag[ni], (short)0, acc[mi][ni],
            false, false);
    p ^= 1;
  }

  // ---- epilogue: fold step, BN, (residual,) ReLU ----
  const float step = __uint_as_float(*absbits) * (1.0f / 127.0f);
#pragma unroll
  for (int mi = 0; mi < 4; ++mi) {
#pragma unroll
    for (int r = 0; r < 8; ++r) {
      // C/D layout: VGPR r -> M=r (lanes 0-15) / M=r+8 (lanes 16-31); N=lane%16
      const int m    = m0 + wm * 64 + mi * 16 + half * 8 + r;
      const float iv = gamma[m] * rsqrtf(var[m] + 1e-5f);
      const float bb = beta[m] - mean[m] * iv;
#pragma unroll
      for (int ni = 0; ni < 2; ++ni) {
        const int n = n0 + wn * 32 + ni * 16 + lm;
        float v = acc[mi][ni][r] * step;
        v = v * iv + bb;
        if (STAGE == 1) {
          v = fmaxf(v, 0.0f);
          out_h[(size_t)m * NPOS + n] = (_Float16)v;
        } else {
          const int b  = n / HW;
          const int yx = n - b * HW;
          const size_t gi = ((size_t)b * CCH + m) * HW + yx;  // NCHW
          v += resid[gi];
          v = fmaxf(v, 0.0f);
          out_f[gi] = v;
        }
      }
    }
  }
}

// ---------------------------------------------------------------------------
// Workspace layout (bytes), total ~53.8 MB:
//   [0,8)   absmax bits for w1, w2
//   +256    wq1 f16 levels [256][2304]; then wq2; then xh f16 [C][N];
//   then o1h f16 [C][N].
// ---------------------------------------------------------------------------
extern "C" void kernel_launch(void* const* d_in, const int* in_sizes, int n_in,
                              void* d_out, int out_size, void* d_ws,
                              size_t ws_size, hipStream_t stream) {
  (void)in_sizes; (void)n_in; (void)out_size; (void)ws_size;
  const float* x  = (const float*)d_in[0];
  const float* w1 = (const float*)d_in[1];
  const float* g1 = (const float*)d_in[2];
  const float* b1 = (const float*)d_in[3];
  const float* m1 = (const float*)d_in[4];
  const float* v1 = (const float*)d_in[5];
  const float* w2 = (const float*)d_in[6];
  const float* g2 = (const float*)d_in[7];
  const float* b2 = (const float*)d_in[8];
  const float* m2 = (const float*)d_in[9];
  const float* v2 = (const float*)d_in[10];
  float* out = (float*)d_out;

  unsigned* absbits = (unsigned*)d_ws;
  _Float16* wq1 = (_Float16*)((char*)d_ws + 256);
  _Float16* wq2 = wq1 + (size_t)CCH * KDIM;
  _Float16* xh  = wq2 + (size_t)CCH * KDIM;
  _Float16* o1h = xh  + (size_t)CCH * NPOS;

  init_absmax<<<1, 32, 0, stream>>>(absbits);
  absmax_kernel<<<512, 256, 0, stream>>>(w1, absbits + 0, WELEMS);
  absmax_kernel<<<512, 256, 0, stream>>>(w2, absbits + 1, WELEMS);
  quant_kernel<<<(CCH * CCH + 255) / 256, 256, 0, stream>>>(w1, absbits + 0, wq1);
  quant_kernel<<<(CCH * CCH + 255) / 256, 256, 0, stream>>>(w2, absbits + 1, wq2);
  nchw_to_cn_f16<<<(BATCH * CCH * HW + 255) / 256, 256, 0, stream>>>(x, xh);

  dim3 grid(NPOS / TILE_N, CCH / TILE_M);  // 392 x 2
  conv_bn_wmma<1><<<grid, 256, 0, stream>>>(wq1, xh, absbits + 0,
                                            g1, b1, m1, v1,
                                            nullptr, o1h, nullptr);
  conv_bn_wmma<2><<<grid, 256, 0, stream>>>(wq2, o1h, absbits + 1,
                                            g2, b2, m2, v2,
                                            x, nullptr, out);
}
